// CausalSelfAttention_33260226740786
// MI455X (gfx1250) — matrix operand order repaired and checked by reading, not compiled, not run on hardware
//
#include <hip/hip_runtime.h>
#include <hip/hip_bf16.h>

// ---------------------------------------------------------------------------
// CausalSelfAttention for MI455X (gfx1250), wave32 + V_WMMA_F32_16X16X32_BF16
// B=4, T=2048, C=1024, H=16, HD=64
// GEMMs: LDS-staged, double-buffered weight tiles via global_load_async_to_lds
// ---------------------------------------------------------------------------

typedef __bf16  v16bf __attribute__((ext_vector_type(16)));
typedef float   v8f   __attribute__((ext_vector_type(8)));
typedef int     v4i   __attribute__((ext_vector_type(4)));

static constexpr int Bc = 4;
static constexpr int Tc = 2048;
static constexpr int Cc = 1024;
static constexpr int Hc = 16;
static constexpr int HDc = 64;
static constexpr int BHc = Bc * Hc;          // 64
static constexpr int Mrows = Bc * Tc;        // 8192

union BF16Frag {
    v16bf v;
    uint4 q[2];
    unsigned short u[16];
};

__device__ __forceinline__ unsigned short f2bf(float f) {
    unsigned int u = __builtin_bit_cast(unsigned int, f);
    u += 0x7FFFu + ((u >> 16) & 1u);            // round-to-nearest-even
    return (unsigned short)(u >> 16);
}

// ---- CDNA5 async global->LDS staging (guarded; sync fallback) --------------
#if defined(__HIP_DEVICE_COMPILE__) && \
    __has_builtin(__builtin_amdgcn_global_load_async_to_lds_b128) && \
    __has_builtin(__builtin_amdgcn_s_wait_asynccnt)
#define HAS_ASYNC_LDS 1
#define WAIT_ASYNC(n) __builtin_amdgcn_s_wait_asynccnt(n)
typedef v4i __attribute__((address_space(1))) as1_v4i;
typedef v4i __attribute__((address_space(3))) as3_v4i;
#else
#define HAS_ASYNC_LDS 0
#define WAIT_ASYNC(n)
#endif

__device__ __forceinline__ void stage_b128(const unsigned short* src,
                                           unsigned short* dstLds) {
#if HAS_ASYNC_LDS
    __builtin_amdgcn_global_load_async_to_lds_b128(
        (as1_v4i*)src, (as3_v4i*)dstLds, 0, 0);
#else
    *(uint4*)dstLds = *(const uint4*)src;
#endif
}

// ------------------------------ cast kernels -------------------------------

__global__ void cast_bf16_kernel(const float* __restrict__ in,
                                 unsigned short* __restrict__ out, int n) {
    int i = blockIdx.x * blockDim.x + threadIdx.x;
    if (i < n) out[i] = f2bf(in[i]);
}

// in: [K,N] row-major f32  ->  out: [N,K] bf16 (transposed)
__global__ void cast_transpose_kernel(const float* __restrict__ in,
                                      unsigned short* __restrict__ out,
                                      int K, int N) {
    int i = blockIdx.x * blockDim.x + threadIdx.x;
    if (i < K * N) {
        int k = i / N;
        int n = i - k * N;
        out[(size_t)n * K + k] = f2bf(in[i]);
    }
}

// --------------------------- shared GEMM core ------------------------------
// A [M,1024] bf16 row-major; B staged from wT [N,1024] bf16 (N,K layout).
// Block: 256 threads (8 waves), block tile 256 rows x 64 cols.
// Wave tile: 32 rows x 64 cols (2 A-frags, 8 accumulators).
// LDS: double-buffered 64x32 B tile, stride 40 halfs (bank-conflict-free).
// All 4 B-fragments preloaded into registers -> one s_wait_dscnt, 8 WMMAs.

#define GEMM_CORE(XB, WT, ACC, SB)                                            \
    const int lane = threadIdx.x & 31;                                        \
    const int wave = threadIdx.x >> 5;                                        \
    const int ln = lane & 15, hg = lane >> 4;                                 \
    const int mBase = blockIdx.x * 256 + wave * 32;                           \
    const int nBase = blockIdx.y * 64;                                        \
    v8f ACC[2][4] = {};                                                       \
    BF16Frag Aa[2], Ab[2];                                                    \
    auto loadA = [&](BF16Frag(&A)[2], int kk) {                               \
        _Pragma("unroll") for (int i = 0; i < 2; ++i) {                       \
            const unsigned short* p =                                         \
                XB + (size_t)(mBase + i * 16 + ln) * Cc + kk + hg * 8;        \
            A[i].q[0] = *(const uint4*)(p);                                   \
            A[i].q[1] = *(const uint4*)(p + 16);                              \
        }                                                                     \
    };                                                                        \
    auto stageB = [&](int buf, int kk) {                                      \
        const int tid = threadIdx.x;                                          \
        const int n = tid >> 2, k8 = (tid & 3) * 8;                           \
        const unsigned short* src = WT + (size_t)(nBase + n) * Cc + kk + k8;  \
        stage_b128(src, &SB[buf][n * 40 + k8]);                               \
    };                                                                        \
    auto phase = [&](BF16Frag(&Acur)[2], BF16Frag(&Anext)[2], int buf,        \
                     int kk) {                                                \
        const bool hasNext = (kk + 32) < Cc;                                  \
        if (hasNext) {                                                        \
            stageB(buf ^ 1, kk + 32);                                         \
            loadA(Anext, kk + 32);                                            \
        }                                                                     \
        if (hasNext) { WAIT_ASYNC(1); } else { WAIT_ASYNC(0); }               \
        __syncthreads();                                                      \
        BF16Frag Bf[4];                                                       \
        _Pragma("unroll") for (int c = 0; c < 4; ++c) {                       \
            const unsigned short* bp = &SB[buf][(c * 16 + ln) * 40 + hg * 8];\
            Bf[c].q[0] = *(const uint4*)(bp);                                 \
            Bf[c].q[1] = *(const uint4*)(bp + 16);                            \
        }                                                                     \
        _Pragma("unroll") for (int c = 0; c < 4; ++c) {                       \
            ACC[0][c] = __builtin_amdgcn_wmma_f32_16x16x32_bf16(              \
                false, Acur[0].v, false, Bf[c].v, (short)0, ACC[0][c], false, \
                false);                                                       \
            ACC[1][c] = __builtin_amdgcn_wmma_f32_16x16x32_bf16(              \
                false, Acur[1].v, false, Bf[c].v, (short)0, ACC[1][c], false, \
                false);                                                       \
        }                                                                     \
        __syncthreads();                                                      \
    };                                                                        \
    loadA(Aa, 0);                                                             \
    stageB(0, 0);                                                             \
    for (int kk = 0; kk < Cc; kk += 64) {                                     \
        phase(Aa, Ab, 0, kk);                                                 \
        phase(Ab, Aa, 1, kk + 32);                                            \
    }

// ------------------------------ QKV GEMM -----------------------------------

__global__ __launch_bounds__(256)
void gemm_qkv_kernel(const unsigned short* __restrict__ xb,
                     const unsigned short* __restrict__ wT,
                     unsigned short* __restrict__ Qb,
                     unsigned short* __restrict__ Kb,
                     unsigned short* __restrict__ Vt) {
    __shared__ unsigned short sB[2][64 * 40];
    GEMM_CORE(xb, wT, acc, sB)

#pragma unroll
    for (int i = 0; i < 2; ++i) {
#pragma unroll
        for (int c = 0; c < 4; ++c) {
            const int nn = nBase + c * 16 + ln;
            const int which = nn >> 10;         // 0=Q 1=K 2=V
            const int cc = nn & 1023;
            const int h = cc >> 6, d = cc & 63;
#pragma unroll
            for (int r = 0; r < 8; ++r) {
                const int m = mBase + i * 16 + r + hg * 8;
                const int b = m >> 11, t = m & 2047;
                const int bh = b * Hc + h;
                const unsigned short bv = f2bf(acc[i][c][r]);
                if (which == 0)      Qb[((size_t)bh * Tc + t) * HDc + d] = bv;
                else if (which == 1) Kb[((size_t)bh * Tc + t) * HDc + d] = bv;
                else                 Vt[((size_t)bh * HDc + d) * Tc + t] = bv;
            }
        }
    }
}

// ------------------------------ proj GEMM ----------------------------------

__global__ __launch_bounds__(256)
void gemm_proj_kernel(const unsigned short* __restrict__ yb,
                      const unsigned short* __restrict__ wT,
                      const float* __restrict__ bias,
                      float* __restrict__ out) {
    __shared__ unsigned short sB[2][64 * 40];
    GEMM_CORE(yb, wT, acc, sB)

#pragma unroll
    for (int c = 0; c < 4; ++c) {
        const int nn = nBase + c * 16 + ln;
        const float bv = bias[nn];
#pragma unroll
        for (int i = 0; i < 2; ++i) {
#pragma unroll
            for (int r = 0; r < 8; ++r) {
                const int m = mBase + i * 16 + r + hg * 8;
                out[(size_t)m * Cc + nn] = acc[i][c][r] + bv;
            }
        }
    }
}

// --------------------------- flash attention -------------------------------
// One wave per (b,h, 16-row Q tile). Online softmax, causal.
// Q: [BH,T,HD] bf16, K: [BH,T,HD] bf16, Vt: [BH,HD,T] bf16, Y: [B,T,C] bf16.

__global__ __launch_bounds__(32)
void attn_kernel(const unsigned short* __restrict__ Qb,
                 const unsigned short* __restrict__ Kb,
                 const unsigned short* __restrict__ Vt,
                 unsigned short* __restrict__ Yb) {
    __shared__ unsigned short sP[16 * 40];      // 16x32 P tile, stride 40 halfs

    const int lane = threadIdx.x & 31;
    const int ln = lane & 15, hg = lane >> 4;
    const int tile  = blockIdx.x;               // 0 .. BH*(T/16)-1
    const int bh    = tile >> 7;                // / (T/16)
    const int qt    = tile & 127;
    const int qbase = qt * 16;
    const int b = bh >> 4, h = bh & 15;

    // Q A-fragments for the two 32-wide K-dim steps over HD=64
    BF16Frag QA[2];
    {
        const unsigned short* qp = Qb + ((size_t)bh * Tc + qbase + ln) * HDc;
#pragma unroll
        for (int s = 0; s < 2; ++s) {
            const unsigned short* p = qp + s * 32 + hg * 8;
            QA[s].q[0] = *(const uint4*)(p);
            QA[s].q[1] = *(const uint4*)(p + 16);
        }
    }

    v8f O[4] = {};
    float mrow[8], lrow[8];
#pragma unroll
    for (int r = 0; r < 8; ++r) { mrow[r] = -3.0e38f; lrow[r] = 0.f; }

    const unsigned short* kBase = Kb + (size_t)bh * Tc * HDc;
    const unsigned short* vBase = Vt + (size_t)bh * HDc * Tc;

    const int nkb = (qbase + 16 + 31) >> 5;     // causal bound, 32 keys/iter
    for (int it = 0; it < nkb; ++it) {
        const int kb = it * 32;

        // ---- preload all K fragments, then S = (Q K^T) ---------------------
        BF16Frag KB[2][2];
#pragma unroll
        for (int c = 0; c < 2; ++c) {
#pragma unroll
            for (int s = 0; s < 2; ++s) {
                const unsigned short* kp =
                    kBase + (size_t)(kb + c * 16 + ln) * HDc + s * 32 + hg * 8;
                KB[c][s].q[0] = *(const uint4*)(kp);
                KB[c][s].q[1] = *(const uint4*)(kp + 16);
            }
        }
        v8f S[2] = {};
#pragma unroll
        for (int c = 0; c < 2; ++c)
#pragma unroll
            for (int s = 0; s < 2; ++s)
                S[c] = __builtin_amdgcn_wmma_f32_16x16x32_bf16(
                    false, QA[s].v, false, KB[c][s].v, (short)0, S[c],
                    false, false);

        // ---- scale + causal mask ------------------------------------------
#pragma unroll
        for (int c = 0; c < 2; ++c) {
#pragma unroll
            for (int r = 0; r < 8; ++r) {
                const int tq = qbase + r + hg * 8;
                const int tk = kb + c * 16 + ln;
                const float v = S[c][r] * 0.125f;   // HD^-0.5
                S[c][r] = (tk > tq) ? -3.0e38f : v;
            }
        }

        // ---- online softmax (row stats across 16-lane groups) -------------
        float alpha[8];
#pragma unroll
        for (int r = 0; r < 8; ++r) {
            float rm = fmaxf(S[0][r], S[1][r]);
            for (int msk = 1; msk < 16; msk <<= 1)
                rm = fmaxf(rm, __shfl_xor(rm, msk, 32));
            const float mnew = fmaxf(mrow[r], rm);
            alpha[r] = __expf(mrow[r] - mnew);
            mrow[r] = mnew;
            const float p0 = __expf(S[0][r] - mnew);
            const float p1 = __expf(S[1][r] - mnew);
            S[0][r] = p0; S[1][r] = p1;
            float rs = p0 + p1;
            for (int msk = 1; msk < 16; msk <<= 1)
                rs += __shfl_xor(rs, msk, 32);
            lrow[r] = lrow[r] * alpha[r] + rs;
        }
#pragma unroll
        for (int c = 0; c < 4; ++c)
#pragma unroll
            for (int r = 0; r < 8; ++r) O[c][r] *= alpha[r];

        // ---- P: C-layout -> A-layout via LDS ------------------------------
        __syncthreads();
#pragma unroll
        for (int r = 0; r < 8; ++r) {
            const int m = r + hg * 8;
            sP[m * 40 + ln]      = f2bf(S[0][r]);
            sP[m * 40 + 16 + ln] = f2bf(S[1][r]);
        }
        __syncthreads();
        BF16Frag PA;
        {
            const unsigned short* p = sP + ln * 40 + hg * 8;
            PA.q[0] = *(const uint4*)(p);
            PA.q[1] = *(const uint4*)(p + 16);
        }
        __syncthreads();

        // ---- preload all V fragments, then O += P @ V ----------------------
        BF16Frag VB[4];
#pragma unroll
        for (int c2 = 0; c2 < 4; ++c2) {
            const unsigned short* vp =
                vBase + (size_t)(c2 * 16 + ln) * Tc + kb + hg * 8;
            VB[c2].q[0] = *(const uint4*)(vp);
            VB[c2].q[1] = *(const uint4*)(vp + 16);
        }
#pragma unroll
        for (int c2 = 0; c2 < 4; ++c2)
            O[c2] = __builtin_amdgcn_wmma_f32_16x16x32_bf16(
                false, PA.v, false, VB[c2].v, (short)0, O[c2], false, false);
    }

    // ---- normalize + store y[b, t, h*64+d] as bf16 -------------------------
#pragma unroll
    for (int c2 = 0; c2 < 4; ++c2) {
#pragma unroll
        for (int r = 0; r < 8; ++r) {
            const float inv = 1.0f / lrow[r];
            const int m = r + hg * 8;
            const int d = c2 * 16 + ln;
            Yb[((size_t)(b * Tc) + qbase + m) * Cc + h * HDc + d] =
                f2bf(O[c2][r] * inv);
        }
    }
}

// ------------------------------ launcher -----------------------------------

extern "C" void kernel_launch(void* const* d_in, const int* in_sizes, int n_in,
                              void* d_out, int out_size, void* d_ws, size_t ws_size,
                              hipStream_t stream) {
    const float* x      = (const float*)d_in[0];   // [4,2048,1024]
    const float* w_qkv  = (const float*)d_in[1];   // [1024,3072]
    const float* w_proj = (const float*)d_in[2];   // [1024,1024]
    const float* b_proj = (const float*)d_in[3];   // [1024]
    float* out = (float*)d_out;

    unsigned short* ws = (unsigned short*)d_ws;
    const size_t NX   = (size_t)Mrows * Cc;        // 8M
    const size_t NQKV = (size_t)Cc * 3 * Cc;       // 3M
    const size_t NPRJ = (size_t)Cc * Cc;           // 1M
    const size_t NHED = (size_t)BHc * Tc * HDc;    // 8M

    unsigned short* xb     = ws;
    unsigned short* wqkvT  = xb + NX;
    unsigned short* wprojT = wqkvT + NQKV;
    unsigned short* Qb     = wprojT + NPRJ;
    unsigned short* Kb     = Qb + NHED;
    unsigned short* Vt     = Kb + NHED;
    unsigned short* Yb     = Vt + NHED;

    cast_bf16_kernel<<<(int)((NX + 255) / 256), 256, 0, stream>>>(x, xb, (int)NX);
    cast_transpose_kernel<<<(int)((NQKV + 255) / 256), 256, 0, stream>>>(
        w_qkv, wqkvT, Cc, 3 * Cc);
    cast_transpose_kernel<<<(int)((NPRJ + 255) / 256), 256, 0, stream>>>(
        w_proj, wprojT, Cc, Cc);

    gemm_qkv_kernel<<<dim3(Mrows / 256, (3 * Cc) / 64), 256, 0, stream>>>(
        xb, wqkvT, Qb, Kb, Vt);

    attn_kernel<<<BHc * (Tc / 16), 32, 0, stream>>>(Qb, Kb, Vt, Yb);

    gemm_proj_kernel<<<dim3(Mrows / 256, Cc / 64), 256, 0, stream>>>(
        Yb, wprojT, b_proj, out);
}
